// GCNModel_55860344652096
// MI455X (gfx1250) — matrix operand order repaired and checked by reading, not compiled, not run on hardware
//
#include <hip/hip_runtime.h>
#include <hip/hip_bf16.h>

typedef __attribute__((ext_vector_type(16))) _Float16 v16h;
typedef __attribute__((ext_vector_type(8)))  float    v8f;

#define HIDDEN  64
#define NGRAPH  64
#define NFEAT   128

__device__ __forceinline__ void atomAddF(float* p, float v) {
  (void)__hip_atomic_fetch_add(p, v, __ATOMIC_RELAXED, __HIP_MEMORY_SCOPE_AGENT);
}

// ---------------------------------------------------------------- utilities
__global__ void zero_u32_kernel(unsigned* __restrict__ p, int n) {
  int t = blockIdx.x * blockDim.x + threadIdx.x;
  if (t < n) p[t] = 0u;
}

__global__ void degree_kernel(const int* __restrict__ dst, unsigned* __restrict__ degc, int nE) {
  int e = blockIdx.x * blockDim.x + threadIdx.x;
  if (e < nE) atomicAdd(&degc[dst[e]], 1u);
}

__global__ void dis_kernel(const unsigned* __restrict__ degc, float* __restrict__ dis, int n) {
  int i = blockIdx.x * blockDim.x + threadIdx.x;
  if (i < n) dis[i] = rsqrtf(1.0f + (float)degc[i]);  // +1 for the self loop
}

// ---------------------------------------------------------------- WMMA GEMM
// C[M x 64] = A[M x K] @ B[K x 64], f16 inputs on-the-fly, f32 accumulate.
// One wave per 16x16 output tile; block = (32,4): 4 col-tiles share A rows.
// A-frag (16x32 f16, ISA 7.12.2): lanes 0-15 -> M=lane, halves i map to
//   K = i (i<8) / i+8 (i>=8); lanes 16-31 same rows with K shifted by +8.
// B-frag (32x16 f16): lane holds column col0+lane%16, K order same as A-frag.
// C/D (f32 16x16): element r -> row = r + 8*(lane>=16), col = lane&15.
__global__ void gemm_wmma_kernel(const float* __restrict__ A, const float* __restrict__ B,
                                 float* __restrict__ C, int M, int K) {
  const int lane = threadIdx.x & 31;
  const int half = lane >> 4;          // 0 or 1
  const int l16  = lane & 15;
  const int col0 = (int)threadIdx.y * 16;
  const size_t row0 = (size_t)blockIdx.x * 16;
  if (row0 + 16 > (size_t)M) return;   // M is a multiple of 16 here

  v8f acc = {};
  for (int kb = 0; kb < K; kb += 32) {
    v16h a, b;
    const float* arow = A + (row0 + (size_t)l16) * K + kb + half * 8;
#pragma unroll
    for (int i = 0; i < 8; ++i)  a[i] = (_Float16)arow[i];        // K = base+i
#pragma unroll
    for (int i = 8; i < 16; ++i) a[i] = (_Float16)arow[i + 8];    // K = base+i+8
    const float* brow = B + (size_t)(kb + half * 8) * HIDDEN + col0 + l16;
#pragma unroll
    for (int i = 0; i < 8; ++i)  b[i] = (_Float16)brow[(size_t)i * HIDDEN];
#pragma unroll
    for (int i = 8; i < 16; ++i) b[i] = (_Float16)brow[(size_t)(i + 8) * HIDDEN];
    acc = __builtin_amdgcn_wmma_f32_16x16x32_f16(
        /*neg_a=*/false, a, /*neg_b=*/false, b,
        /*c_mod=*/(short)0, acc, /*reuse_a=*/false, /*reuse_b=*/false);
  }
#pragma unroll
  for (int r = 0; r < 8; ++r)
    C[(row0 + r + half * 8) * HIDDEN + col0 + l16] = acc[r];
}

// ---------------------------------------------------------------- aggregation
// Self-loop term, non-atomic init: agg[i][:] = h[i][:] * dis[i]^2
__global__ void self_init_kernel(const float* __restrict__ h, const float* __restrict__ dis,
                                 float* __restrict__ agg, int n) {
  int t = blockIdx.x * blockDim.x + threadIdx.x;
  if (t >= n * HIDDEN) return;
  float di = dis[t >> 6];
  agg[t] = h[t] * di * di;
}

// One wave32 per edge: each lane moves 2 floats of the 64-float row.
__global__ void edge_agg_kernel(const int* __restrict__ src, const int* __restrict__ dst,
                                const float* __restrict__ dis, const float* __restrict__ h,
                                float* __restrict__ agg, int nE) {
  int e = blockIdx.x * (blockDim.x >> 5) + (threadIdx.x >> 5);
  if (e >= nE) return;
  int lane = threadIdx.x & 31;
  int s = src[e], d = dst[e];
  float norm = dis[s] * dis[d];
  float2 v = ((const float2*)(h + (size_t)s * HIDDEN))[lane];
  float* arow = agg + (size_t)d * HIDDEN + lane * 2;
  atomAddF(arow,     v.x * norm);
  atomAddF(arow + 1, v.y * norm);
}

__global__ void bias_relu_kernel(const float* __restrict__ aggin, const float* __restrict__ bias,
                                 float* __restrict__ out, int n) {
  int t = blockIdx.x * blockDim.x + threadIdx.x;
  if (t >= n * HIDDEN) return;
  out[t] = fmaxf(aggin[t] + bias[t & 63], 0.0f);
}

// ---------------------------------------------------------------- pooling/head
__global__ void pool_kernel(const float* __restrict__ agg, const float* __restrict__ bias,
                            const int* __restrict__ batch, float* __restrict__ pooled,
                            unsigned* __restrict__ counts, int n) {
  int t = blockIdx.x * blockDim.x + threadIdx.x;
  if (t >= n * HIDDEN) return;
  int i = t >> 6, j = t & 63;
  int g = batch[i];
  float v = fmaxf(agg[t] + bias[j], 0.0f);
  atomAddF(&pooled[g * HIDDEN + j], v);
  if (j == 0) atomicAdd(&counts[g], 1u);
}

__global__ void head_kernel(const float* __restrict__ pooled, const unsigned* __restrict__ counts,
                            const float* __restrict__ Wh, const float* __restrict__ bh,
                            float* __restrict__ out) {
  int g = threadIdx.x;
  if (g >= NGRAPH) return;
  float inv = 1.0f / fmaxf((float)counts[g], 1.0f);
  float s = 0.0f;
#pragma unroll
  for (int j = 0; j < HIDDEN; ++j) s += pooled[g * HIDDEN + j] * inv * Wh[j];
  out[g] = s + bh[0];
}

// ---------------------------------------------------------------- launcher
extern "C" void kernel_launch(void* const* d_in, const int* in_sizes, int n_in,
                              void* d_out, int out_size, void* d_ws, size_t ws_size,
                              hipStream_t stream) {
  const float* x     = (const float*)d_in[0];
  const int*   ei    = (const int*)  d_in[1];   // [2, E]
  const int*   batch = (const int*)  d_in[2];   // [N]
  const float* W1    = (const float*)d_in[3];
  const float* b1    = (const float*)d_in[4];
  const float* W2    = (const float*)d_in[5];
  const float* b2    = (const float*)d_in[6];
  const float* Wh    = (const float*)d_in[7];
  const float* bh    = (const float*)d_in[8];
  float*       out   = (float*)d_out;

  const int N = in_sizes[0] / NFEAT;   // 100000 (multiple of 16)
  const int E = in_sizes[1] / 2;       // 1600000
  const int* src = ei;
  const int* dst = ei + E;

  // workspace layout (ws 256B-aligned; offsets are 64B-aligned since N % 16 == 0)
  char* ws = (char*)d_ws;
  float*    dis    = (float*)ws;                               // N f32
  unsigned* degc   = (unsigned*)(ws + (size_t)N * 4);          // N u32
  float*    bufA   = (float*)(ws + (size_t)N * 8);             // N*64 f32
  float*    bufB   = bufA + (size_t)N * HIDDEN;                // N*64 f32
  float*    pooled = bufB + (size_t)N * HIDDEN;                // 64*64 f32
  unsigned* counts = (unsigned*)(pooled + NGRAPH * HIDDEN);    // 64 u32

  const int TB = 256;
  const int nNH   = N * HIDDEN;
  const int gNH   = (nNH + TB - 1) / TB;
  const int gN    = (N + TB - 1) / TB;
  const int gE    = (E + TB - 1) / TB;
  const int gEdge = (E + 7) / 8;                 // 8 waves of 32 per block

  // degrees -> dis = 1/sqrt(deg+1)
  zero_u32_kernel<<<gN, TB, 0, stream>>>(degc, N);
  degree_kernel<<<gE, TB, 0, stream>>>(dst, degc, E);
  dis_kernel<<<gN, TB, 0, stream>>>(degc, dis, N);

  // ----- layer 1: h = relu(Agg(x @ W1) + b1)
  gemm_wmma_kernel<<<N / 16, dim3(32, 4), 0, stream>>>(x, W1, bufA, N, NFEAT);
  self_init_kernel<<<gNH, TB, 0, stream>>>(bufA, dis, bufB, N);
  edge_agg_kernel<<<gEdge, TB, 0, stream>>>(src, dst, dis, bufA, bufB, E);
  bias_relu_kernel<<<gNH, TB, 0, stream>>>(bufB, b1, bufA, N);

  // ----- layer 2: h = relu(Agg(h @ W2) + b2)   (relu fused into pool)
  gemm_wmma_kernel<<<N / 16, dim3(32, 4), 0, stream>>>(bufA, W2, bufB, N, HIDDEN);
  self_init_kernel<<<gNH, TB, 0, stream>>>(bufB, dis, bufA, N);
  edge_agg_kernel<<<gEdge, TB, 0, stream>>>(src, dst, dis, bufB, bufA, E);

  // ----- mean pool + head
  const int poolWords = NGRAPH * HIDDEN + NGRAPH;
  zero_u32_kernel<<<(poolWords + TB - 1) / TB, TB, 0, stream>>>((unsigned*)pooled, poolWords);
  pool_kernel<<<gNH, TB, 0, stream>>>(bufA, b2, batch, pooled, counts, N);
  head_kernel<<<1, NGRAPH, 0, stream>>>(pooled, counts, Wh, bh, out);
}
